// TpnTaskLoss_33337536151700
// MI455X (gfx1250) — compile-verified, matrix operand order checked by reading, not run on hardware
//
#include <hip/hip_runtime.h>

#define C_CLASSES 64
#define DIM 256
#define ROWSTRIDE (DIM + 8)   // bf16 elems; 264*2=528B row stride -> banks tiled evenly

typedef __attribute__((ext_vector_type(16))) __bf16 v16bf;
typedef __attribute__((ext_vector_type(8)))  __bf16 v8bf;
typedef __attribute__((ext_vector_type(8)))  float  v8f;

// ---------------- workspace layout (floats unless noted) ----------------
// [0)                     sum_s  : 64*256 f32
// [64*256)                sum_t  : 64*256 f32
// [2*64*256)              cnt_s  : 64 f32
// [2*64*256+64)           cnt_t  : 64 f32
// after that (16B aligned): Ubf   : 3 * 64*256 __bf16  (u_s, u_t, u_st)

__global__ void zero_kernel(float* __restrict__ ws, float* __restrict__ out) {
    int idx = blockIdx.x * blockDim.x + threadIdx.x;
    const int total = 2 * C_CLASSES * DIM + 2 * C_CLASSES;
    if (idx < total) ws[idx] = 0.0f;
    if (idx == 0) out[0] = 0.0f;
}

// blockDim.x must be 256 (== DIM). Thread t owns dimension t -> race-free LDS adds.
__global__ void __launch_bounds__(256)
seg_sum_kernel(const float* __restrict__ feat, const long long* __restrict__ label,
               float* __restrict__ gsum, float* __restrict__ gcnt, int N) {
    __shared__ float s_acc[C_CLASSES * DIM];   // 64 KB of the 320 KB WGP LDS
    __shared__ float s_cnt[C_CLASSES];
    const int t = threadIdx.x;
    for (int i = t; i < C_CLASSES * DIM; i += blockDim.x) s_acc[i] = 0.0f;
    if (t < C_CLASSES) s_cnt[t] = 0.0f;
    __syncthreads();

    int chunk = (N + gridDim.x - 1) / gridDim.x;
    int r0 = blockIdx.x * chunk;
    int r1 = min(N, r0 + chunk);
    for (int r = r0; r < r1; ++r) {
        int c = (int)label[r];
        s_acc[c * DIM + t] += feat[(size_t)r * DIM + t];
        if (t == 0) s_cnt[c] += 1.0f;
    }
    __syncthreads();
    for (int c = 0; c < C_CLASSES; ++c)
        atomicAdd(&gsum[c * DIM + t], s_acc[c * DIM + t]);
    if (t < C_CLASSES) atomicAdd(&gcnt[t], s_cnt[t]);
}

__global__ void __launch_bounds__(256)
centroid_kernel(const float* __restrict__ sum_s, const float* __restrict__ sum_t,
                const float* __restrict__ cnt_s, const float* __restrict__ cnt_t,
                __bf16* __restrict__ u_s, __bf16* __restrict__ u_t,
                __bf16* __restrict__ u_st) {
    int idx = blockIdx.x * blockDim.x + threadIdx.x;   // 64*256 elements
    int c = idx >> 8;
    float cs = cnt_s[c], ct = cnt_t[c];
    float ss = sum_s[idx], st = sum_t[idx];
    u_s[idx]  = (__bf16)(ss / cs);
    u_t[idx]  = (__bf16)(st / ct);
    u_st[idx] = (__bf16)((ss + st) / (cs + ct));
}

// Fused: P_{s,t,st} = feats @ U^T via v_wmma_f32_16x16x32_bf16 with the 96KB
// centroid set staged in LDS once per block, then the symmetric-KL epilogue.
// Each wave grid-strides over 16-row tiles (all 64 classes x 3 matrices).
__global__ void __launch_bounds__(256)
tpn_loss_kernel(const float* __restrict__ src, const float* __restrict__ trg,
                const __bf16* __restrict__ Ubf, float* __restrict__ out,
                int N, float scale) {
    __shared__ __bf16 sU[3 * C_CLASSES * ROWSTRIDE];   // ~99 KB of LDS

    const int lane = threadIdx.x & 31;
    const int wave = threadIdx.x >> 5;
    const int half = lane >> 4;     // which 16-lane half
    const int l    = lane & 15;

    // ---- stage centroids into LDS (padded rows), 16B-vector copies ----
    for (int chunk = threadIdx.x; chunk < 3 * C_CLASSES * (DIM / 8);
         chunk += blockDim.x) {
        int row  = chunk / (DIM / 8);    // 0..191  (mat*64 + class)
        int col8 = chunk % (DIM / 8);
        v8bf v = *(const v8bf*)(Ubf + (size_t)row * DIM + col8 * 8);
        *(v8bf*)(sU + (size_t)row * ROWSTRIDE + col8 * 8) = v;
    }
    __syncthreads();

    const long totalRows = 2L * (long)N;
    const long numTiles  = totalRows / 16;
    const long waveStride = (long)gridDim.x * (blockDim.x / 32);

    float local = 0.0f;

    for (long tile = (long)blockIdx.x * (blockDim.x / 32) + wave;
         tile < numTiles; tile += waveStride) {
        // Compiler memory barrier: forbid LICM from hoisting the (loop-
        // invariant) LDS B-fragment loads out of the tile loop — hoisting
        // needs ~768 VGPRs and spills every fragment to scratch (seen in
        // round-2 codegen). Re-reading LDS each tile is the intended, cheap
        // behavior.
        asm volatile("" ::: "memory");

        const long tileRow = tile * 16;

        // A-fragment row for this lane (rows 0..N-1 = src, N..2N-1 = trg)
        long arow = tileRow + l;
        const float* rowptr = (arow < (long)N) ? (src + (size_t)arow * DIM)
                                               : (trg + (size_t)(arow - N) * DIM);

        const v8f vzero = {0.f, 0.f, 0.f, 0.f, 0.f, 0.f, 0.f, 0.f};
        v8f acc[3][4];
        #pragma unroll
        for (int m = 0; m < 3; ++m)
            #pragma unroll
            for (int nb = 0; nb < 4; ++nb) acc[m][nb] = vzero;

        #pragma unroll
        for (int k0 = 0; k0 < DIM; k0 += 32) {
            // ISA 16-bit A layout: lane<16 -> K {k0+0..7, k0+16..23}
            //                      lane>=16 -> K {k0+8..15, k0+24..31}
            const int kb1 = k0 + half * 8;
            const int kb2 = k0 + 16 + half * 8;

            float4 f0 = *(const float4*)(rowptr + kb1);
            float4 f1 = *(const float4*)(rowptr + kb1 + 4);
            float4 f2 = *(const float4*)(rowptr + kb2);
            float4 f3 = *(const float4*)(rowptr + kb2 + 4);
            v16bf a;
            a[0]=(__bf16)f0.x;  a[1]=(__bf16)f0.y;  a[2]=(__bf16)f0.z;  a[3]=(__bf16)f0.w;
            a[4]=(__bf16)f1.x;  a[5]=(__bf16)f1.y;  a[6]=(__bf16)f1.z;  a[7]=(__bf16)f1.w;
            a[8]=(__bf16)f2.x;  a[9]=(__bf16)f2.y;  a[10]=(__bf16)f2.z; a[11]=(__bf16)f2.w;
            a[12]=(__bf16)f3.x; a[13]=(__bf16)f3.y; a[14]=(__bf16)f3.z; a[15]=(__bf16)f3.w;

            #pragma unroll
            for (int m = 0; m < 3; ++m) {
                #pragma unroll
                for (int nb = 0; nb < 4; ++nb) {
                    // B = U^T slice: b[k][n] = U[nb*16 + n][k]; mirrors A layout
                    const __bf16* up =
                        sU + (size_t)(m * C_CLASSES + nb * 16 + l) * ROWSTRIDE;
                    v8bf b0 = *(const v8bf*)(up + kb1);
                    v8bf b1 = *(const v8bf*)(up + kb2);
                    v16bf b;
                    #pragma unroll
                    for (int i = 0; i < 8; ++i) { b[i] = b0[i]; b[i + 8] = b1[i]; }
                    acc[m][nb] = __builtin_amdgcn_wmma_f32_16x16x32_bf16(
                        false, a, false, b, (short)0, acc[m][nb], false, false);
                }
            }
        }

        // Epilogue. C/D layout: lane<16 holds rows 0..7 (vgpr j = row j),
        // lane>=16 holds rows 8..15; column = nb*16 + l. Row reductions are
        // across the 16 lanes of one half -> shfl_xor offsets 1,2,4,8.
        #pragma unroll
        for (int j = 0; j < 8; ++j) {
            float va[4], vb[4], vc[4];
            #pragma unroll
            for (int nb = 0; nb < 4; ++nb) {
                va[nb] = acc[0][nb][j];
                vb[nb] = acc[1][nb][j];
                vc[nb] = acc[2][nb][j];
            }
            float ma = fmaxf(fmaxf(va[0], va[1]), fmaxf(va[2], va[3]));
            float mb = fmaxf(fmaxf(vb[0], vb[1]), fmaxf(vb[2], vb[3]));
            float mc = fmaxf(fmaxf(vc[0], vc[1]), fmaxf(vc[2], vc[3]));
            #pragma unroll
            for (int off = 1; off <= 8; off <<= 1) {
                ma = fmaxf(ma, __shfl_xor(ma, off, 32));
                mb = fmaxf(mb, __shfl_xor(mb, off, 32));
                mc = fmaxf(mc, __shfl_xor(mc, off, 32));
            }
            float ea = 0.f, eb = 0.f, ec = 0.f;
            #pragma unroll
            for (int i = 0; i < 4; ++i) {
                ea += __expf(va[i] - ma);
                eb += __expf(vb[i] - mb);
                ec += __expf(vc[i] - mc);
            }
            #pragma unroll
            for (int off = 1; off <= 8; off <<= 1) {
                ea += __shfl_xor(ea, off, 32);
                eb += __shfl_xor(eb, off, 32);
                ec += __shfl_xor(ec, off, 32);
            }
            float base_a = ma + __logf(ea);
            float base_b = mb + __logf(eb);
            float base_c = mc + __logf(ec);
            #pragma unroll
            for (int i = 0; i < 4; ++i) {
                float la = va[i] - base_a;
                float lb = vb[i] - base_b;
                float lc = vc[i] - base_c;
                float pa = __expf(la), pb = __expf(lb), pc = __expf(lc);
                // pb(lb-la)+pa(la-lb) + pc(lc-la)+pa(la-lc) + pc(lc-lb)+pb(lb-lc)
                local += pa * (2.f * la - lb - lc)
                       + pb * (2.f * lb - la - lc)
                       + pc * (2.f * lc - la - lb);
            }
        }
    }

    // wave reduce, one atomic per wave
    #pragma unroll
    for (int off = 16; off >= 1; off >>= 1) local += __shfl_xor(local, off, 32);
    if (lane == 0) atomicAdd(out, local * scale);
}

extern "C" void kernel_launch(void* const* d_in, const int* in_sizes, int n_in,
                              void* d_out, int out_size, void* d_ws, size_t ws_size,
                              hipStream_t stream) {
    const float*     src  = (const float*)d_in[0];
    const float*     trg  = (const float*)d_in[1];
    const long long* slab = (const long long*)d_in[2];
    const long long* tlab = (const long long*)d_in[3];
    // d_in[4] (trg_feat_un) unused by the reference loss value.
    const int N = in_sizes[2];

    float* out = (float*)d_out;
    float* ws  = (float*)d_ws;
    float* sum_s = ws;
    float* sum_t = ws + C_CLASSES * DIM;
    float* cnt_s = ws + 2 * C_CLASSES * DIM;
    float* cnt_t = cnt_s + C_CLASSES;
    __bf16* Ubf  = (__bf16*)(cnt_t + C_CLASSES);   // 3 * 64*256 bf16, 16B aligned

    // 1) zero sums/counts + output
    const int zelems = 2 * C_CLASSES * DIM + 2 * C_CLASSES;
    zero_kernel<<<(zelems + 255) / 256, 256, 0, stream>>>(ws, out);

    // 2) per-class sums & counts
    seg_sum_kernel<<<512, 256, 0, stream>>>(src, slab, sum_s, cnt_s, N);
    seg_sum_kernel<<<512, 256, 0, stream>>>(trg, tlab, sum_t, cnt_t, N);

    // 3) centroids (bf16 for the WMMA path)
    centroid_kernel<<<C_CLASSES * DIM / 256, 256, 0, stream>>>(
        sum_s, sum_t, cnt_s, cnt_t,
        Ubf, Ubf + C_CLASSES * DIM, Ubf + 2 * C_CLASSES * DIM);

    // 4) fused GEMM (bf16 WMMA, LDS-staged B) + symmetric KL
    long tiles  = (2L * N + 15) / 16;
    long wantBlocks = (tiles + 7) / 8;             // 8 waves/block
    int  blocks = (int)(wantBlocks < 256 ? wantBlocks : 256);  // grid-stride tiles
    float scale = 1.0f / (6.0f * (float)(2.0 * (double)N) * (float)C_CLASSES);
    tpn_loss_kernel<<<blocks, 256, 0, stream>>>(src, trg, Ubf, out, N, scale);
}
